// RSSM_16054587752464
// MI455X (gfx1250) — compile-verified, matrix operand order checked by reading, not run on hardware
//
#include <hip/hip_runtime.h>

// ---------------------------------------------------------------------------
// RSSM rollout for MI455X (gfx1250): bf16 WMMA (v_wmma_f32_16x16x32_bf16)
// Weights (~43 MB bf16) are L2-resident across all 15 steps -> compute bound.
// Per-wave tiles sized so every A-fragment load feeds >=3 WMMAs.
// ---------------------------------------------------------------------------

typedef __attribute__((ext_vector_type(8)))  float  v8f;
typedef __attribute__((ext_vector_type(8)))  __bf16 v8bf;
typedef __attribute__((ext_vector_type(16))) __bf16 v16bf;

#define B_    1024
#define LC    1024      // L*C
#define IN_P  1056      // In=1030 padded to multiple of 32
#define D_    4096
#define NB    8
#define BS    512       // block size D/NB
#define G3    1536      // 3*BS
#define HID   512
#define FEAT  5120      // D + L*C
#define T_    15

__device__ __forceinline__ unsigned short f2bf(float f) {
  union { float f; unsigned u; } v; v.f = f;
  unsigned u = v.u;
  unsigned r = u + 0x7FFFu + ((u >> 16) & 1u);   // round-to-nearest-even
  return (unsigned short)(r >> 16);
}

// A-frag (16x32, M=lane&15) and B-frag (32x16, N=lane&15) have identical
// per-lane indexing for a [rows][K] row-major bf16 source:
//   kb = k0 + 8*(lane>=16); elems 0..7 = K kb..kb+7 ; elems 8..15 = K kb+16..kb+23
__device__ __forceinline__ v16bf load_frag(const unsigned short* base, int ld,
                                           int t0, int k0, int lane) {
  int r  = t0 + (lane & 15);
  int kb = k0 + ((lane >> 4) << 3);
  const unsigned short* q = base + (size_t)r * ld + kb;
  v8bf lo = *(const v8bf*)q;
  v8bf hi = *(const v8bf*)(q + 16);
  v16bf f;
#pragma unroll
  for (int i = 0; i < 8; ++i) { f[i] = lo[i]; f[i + 8] = hi[i]; }
  return f;
}

__device__ __forceinline__ v8f wmma_bf(v16bf a, v16bf b, v8f c) {
  return __builtin_amdgcn_wmma_f32_16x16x32_bf16(false, a, false, b,
                                                 (short)0, c, false, false);
}

__device__ __forceinline__ float sigm(float x) { return 1.f / (1.f + __expf(-x)); }

// ---------------------------------------------------------------------------
// one-time conversion / init kernels
// ---------------------------------------------------------------------------
__global__ void k_conv_pad(const float* __restrict__ in, unsigned short* __restrict__ out,
                           int rows, int cin, int cout) {
  size_t idx = (size_t)blockIdx.x * blockDim.x + threadIdx.x;
  if (idx >= (size_t)rows * cout) return;
  int r = (int)(idx / cout), c = (int)(idx % cout);
  out[idx] = (c < cin) ? f2bf(in[(size_t)r * cin + c]) : (unsigned short)0;
}

__global__ void k_conv_t(const float* __restrict__ in, unsigned short* __restrict__ out,
                         int K, int N) {   // in [K][N] -> out [N][K]
  size_t idx = (size_t)blockIdx.x * blockDim.x + threadIdx.x;
  if (idx >= (size_t)K * N) return;
  int n = (int)(idx / K), k = (int)(idx % K);
  out[idx] = f2bf(in[(size_t)k * N + n]);
}

__global__ void k_init_x(const float* __restrict__ stoch0, unsigned short* __restrict__ xbf) {
  size_t idx = (size_t)blockIdx.x * blockDim.x + threadIdx.x;
  if (idx >= (size_t)B_ * IN_P) return;
  int b = (int)(idx / IN_P), c = (int)(idx % IN_P);
  xbf[idx] = (c < LC) ? f2bf(stoch0[(size_t)b * LC + c]) : (unsigned short)0;
}

__global__ void k_init_h(const float* __restrict__ deter0, float* __restrict__ deter,
                         unsigned short* __restrict__ hbf) {
  size_t idx = (size_t)blockIdx.x * blockDim.x + threadIdx.x;
  if (idx >= (size_t)B_ * D_) return;
  float v = deter0[idx];
  deter[idx] = v;
  hbf[idx] = f2bf(v);
}

__global__ void k_setact(const int* __restrict__ act_t, unsigned short* __restrict__ xbf) {
  int b = blockIdx.x * blockDim.x + threadIdx.x;
  if (b >= B_) return;
  int a = act_t[b];
#pragma unroll
  for (int j = 0; j < 6; ++j)
    xbf[(size_t)b * IN_P + LC + j] = (j == a) ? (unsigned short)0x3F80 : (unsigned short)0;
}

// ---------------------------------------------------------------------------
// Fused block-GRU: gi = x@W_ih^T, gh = h@W_hh^T (block diag), then gating.
// grid (B/32, NB, BS/128), 256 threads = 8 waves.
// Each wave: 32 rows x 16 units -> 8 accumulators (r,z,gi_n,gh_n for 2 M-tiles)
// Per K-step: 5 fragment loads feed 6 WMMAs.
// ---------------------------------------------------------------------------
__global__ __launch_bounds__(256)
void k_gates_gru(const unsigned short* __restrict__ xbf,     // [B][IN_P]
                 const unsigned short* __restrict__ hbf_in,  // [B][D]
                 const unsigned short* __restrict__ wih,     // [NB*G3][IN_P]
                 const unsigned short* __restrict__ whh,     // [NB*G3][BS]
                 const float* __restrict__ b_ih,             // [NB][G3]
                 const float* __restrict__ b_hh,
                 float* __restrict__ deter,                  // [B][D] in/out
                 unsigned short* __restrict__ hbf_out,       // [B][D]
                 float* __restrict__ out_feat)               // d_out + t*B*FEAT
{
  const int lane = threadIdx.x & 31;
  const int wave = threadIdx.x >> 5;
  const int m0   = blockIdx.x * 32;              // 2 M sub-tiles per wave
  const int kblk = blockIdx.y;
  const int u0   = blockIdx.z * 128 + wave * 16; // 16 units per wave

  v8f ar0, ar1, az0, az1, agin0, agin1, aghn0, aghn1;
#pragma unroll
  for (int i = 0; i < 8; ++i) {
    ar0[i]=0.f; ar1[i]=0.f; az0[i]=0.f; az1[i]=0.f;
    agin0[i]=0.f; agin1[i]=0.f; aghn0[i]=0.f; aghn1[i]=0.f;
  }

  const unsigned short* wih_r = wih + (size_t)(kblk * G3        ) * IN_P;
  const unsigned short* wih_z = wih + (size_t)(kblk * G3 +   BS ) * IN_P;
  const unsigned short* wih_n = wih + (size_t)(kblk * G3 + 2*BS ) * IN_P;
  for (int k0 = 0; k0 < IN_P; k0 += 32) {
    v16bf a0 = load_frag(xbf, IN_P, m0,      k0, lane);
    v16bf a1 = load_frag(xbf, IN_P, m0 + 16, k0, lane);
    v16bf br = load_frag(wih_r, IN_P, u0, k0, lane);
    v16bf bz = load_frag(wih_z, IN_P, u0, k0, lane);
    v16bf bn = load_frag(wih_n, IN_P, u0, k0, lane);
    ar0   = wmma_bf(a0, br, ar0);   ar1   = wmma_bf(a1, br, ar1);
    az0   = wmma_bf(a0, bz, az0);   az1   = wmma_bf(a1, bz, az1);
    agin0 = wmma_bf(a0, bn, agin0); agin1 = wmma_bf(a1, bn, agin1);
  }

  const unsigned short* whh_r = whh + (size_t)(kblk * G3        ) * BS;
  const unsigned short* whh_z = whh + (size_t)(kblk * G3 +   BS ) * BS;
  const unsigned short* whh_n = whh + (size_t)(kblk * G3 + 2*BS ) * BS;
  const unsigned short* hsl   = hbf_in + kblk * BS;   // block slice, row stride D_
  for (int k0 = 0; k0 < BS; k0 += 32) {
    v16bf a0 = load_frag(hsl, D_, m0,      k0, lane);
    v16bf a1 = load_frag(hsl, D_, m0 + 16, k0, lane);
    v16bf br = load_frag(whh_r, BS, u0, k0, lane);
    v16bf bz = load_frag(whh_z, BS, u0, k0, lane);
    v16bf bn = load_frag(whh_n, BS, u0, k0, lane);
    ar0   = wmma_bf(a0, br, ar0);   ar1   = wmma_bf(a1, br, ar1);
    az0   = wmma_bf(a0, bz, az0);   az1   = wmma_bf(a1, bz, az1);
    aghn0 = wmma_bf(a0, bn, aghn0); aghn1 = wmma_bf(a1, bn, aghn1);
  }

  // gating + state update.  C/D layout: n = lane&15, m = vgpr + 8*(lane>=16)
  const int n   = lane & 15;
  const int u   = u0 + n;
  const int col = kblk * BS + u;
  const float bir = b_ih[kblk*G3 + u],        bhr = b_hh[kblk*G3 + u];
  const float biz = b_ih[kblk*G3 + BS + u],   bhz = b_hh[kblk*G3 + BS + u];
  const float bin = b_ih[kblk*G3 + 2*BS + u], bhn = b_hh[kblk*G3 + 2*BS + u];
#pragma unroll
  for (int t2 = 0; t2 < 2; ++t2) {
    v8f Ar = t2 ? ar1 : ar0;
    v8f Az = t2 ? az1 : az0;
    v8f Ai = t2 ? agin1 : agin0;
    v8f Ah = t2 ? aghn1 : aghn0;
    const int mbase = m0 + t2 * 16 + ((lane >> 4) << 3);
#pragma unroll
    for (int i = 0; i < 8; ++i) {
      int b = mbase + i;
      float r  = sigm(Ar[i] + bir + bhr);
      float z  = sigm(Az[i] + biz + bhz);
      float nn = tanhf(Ai[i] + bin + r * (Ah[i] + bhn));
      size_t o = (size_t)b * D_ + col;
      float hold = deter[o];
      float hnew = (1.f - z) * nn + z * hold;
      deter[o]   = hnew;
      hbf_out[o] = f2bf(hnew);
      out_feat[(size_t)b * FEAT + col] = hnew;
    }
  }
}

// ---------------------------------------------------------------------------
// hid = deter@w1 + b1, RMSNorm, *rms_w, SiLU  -> bf16 hid. grid(B/16), 256 thr
// Each wave: 16 rows x 64 cols (1 A-frag : 4 B-frags : 4 WMMAs per K-step).
// ---------------------------------------------------------------------------
__global__ __launch_bounds__(256)
void k_mlp1(const unsigned short* __restrict__ hbf,   // deter_new bf16 [B][D]
            const unsigned short* __restrict__ w1t,   // [HID][D]
            const float* __restrict__ b1, const float* __restrict__ rms_w,
            unsigned short* __restrict__ hidbf)       // [B][HID]
{
  __shared__ float tile[16][HID];
  __shared__ float partial[16][16];
  __shared__ float rrms[16];
  const int lane = threadIdx.x & 31;
  const int wave = threadIdx.x >> 5;
  const int m0   = blockIdx.x * 16;
  const int n0w  = wave * 64;

  v8f acc[4];
#pragma unroll
  for (int j = 0; j < 4; ++j)
#pragma unroll
    for (int i = 0; i < 8; ++i) acc[j][i] = 0.f;

  for (int k0 = 0; k0 < D_; k0 += 32) {
    v16bf a = load_frag(hbf, D_, m0, k0, lane);
#pragma unroll
    for (int j = 0; j < 4; ++j) {
      v16bf b = load_frag(w1t, D_, n0w + j * 16, k0, lane);
      acc[j] = wmma_bf(a, b, acc[j]);
    }
  }
  const int n = lane & 15, mb = (lane >> 4) << 3;
#pragma unroll
  for (int j = 0; j < 4; ++j)
#pragma unroll
    for (int i = 0; i < 8; ++i) {
      int c = n0w + j * 16 + n;
      tile[mb + i][c] = acc[j][i] + b1[c];
    }
  __syncthreads();

  const int row = threadIdx.x >> 4;   // 0..15
  const int c0  = threadIdx.x & 15;
  float s = 0.f;
#pragma unroll
  for (int j = 0; j < 32; ++j) { float v = tile[row][c0 + 16 * j]; s += v * v; }
  partial[row][c0] = s;
  __syncthreads();
  if (threadIdx.x < 16) {
    float t = 0.f;
#pragma unroll
    for (int j = 0; j < 16; ++j) t += partial[threadIdx.x][j];
    rrms[threadIdx.x] = 1.f / sqrtf(t * (1.f / HID) + 1e-6f);
  }
  __syncthreads();
#pragma unroll
  for (int j = 0; j < 32; ++j) {
    int col = c0 + 16 * j;
    float v = tile[row][col] * rms_w[col] * rrms[row];
    v = v * sigm(v);                               // SiLU
    hidbf[(size_t)(m0 + row) * HID + col] = f2bf(v);
  }
}

// ---------------------------------------------------------------------------
// logits = hid@w2 + b2 : M=1024 N=1024 K=512.  grid(B/32, LC/512), 8 waves.
// Each wave: 32 rows x 64 cols (6 fragment loads : 8 WMMAs per K-step).
// ---------------------------------------------------------------------------
__global__ __launch_bounds__(256)
void k_logits(const unsigned short* __restrict__ hidbf,  // [B][HID]
              const unsigned short* __restrict__ w2t,    // [LC][HID]
              const float* __restrict__ b2, float* __restrict__ logits)
{
  const int lane = threadIdx.x & 31;
  const int wave = threadIdx.x >> 5;
  const int m0 = blockIdx.x * 32;
  const int n0 = blockIdx.y * 512 + wave * 64;

  v8f acc[2][4];
#pragma unroll
  for (int t2 = 0; t2 < 2; ++t2)
#pragma unroll
    for (int j = 0; j < 4; ++j)
#pragma unroll
      for (int i = 0; i < 8; ++i) acc[t2][j][i] = 0.f;

  for (int k0 = 0; k0 < HID; k0 += 32) {
    v16bf a0 = load_frag(hidbf, HID, m0,      k0, lane);
    v16bf a1 = load_frag(hidbf, HID, m0 + 16, k0, lane);
#pragma unroll
    for (int j = 0; j < 4; ++j) {
      v16bf b = load_frag(w2t, HID, n0 + j * 16, k0, lane);
      acc[0][j] = wmma_bf(a0, b, acc[0][j]);
      acc[1][j] = wmma_bf(a1, b, acc[1][j]);
    }
  }
  const int nl = lane & 15;
#pragma unroll
  for (int t2 = 0; t2 < 2; ++t2) {
    const int mbase = m0 + t2 * 16 + ((lane >> 4) << 3);
#pragma unroll
    for (int j = 0; j < 4; ++j) {
      int c = n0 + j * 16 + nl;
#pragma unroll
      for (int i = 0; i < 8; ++i)
        logits[(size_t)(mbase + i) * LC + c] = acc[t2][j][i] + b2[c];
    }
  }
}

// ---------------------------------------------------------------------------
// gumbel-softmax straight-through (== hard one-hot of argmax(mix + gumbel))
// ---------------------------------------------------------------------------
__global__ void k_sample(const float* __restrict__ logits, const float* __restrict__ unif_t,
                         float* __restrict__ out_stoch,           // d_out+t*B*FEAT+D_
                         unsigned short* __restrict__ xbf)        // next-step input
{
  int idx = blockIdx.x * blockDim.x + threadIdx.x;
  if (idx >= B_ * 32) return;
  int b = idx >> 5, l = idx & 31;
  const float* lg = logits + (size_t)b * LC + l * 32;
  const float* u  = unif_t + (size_t)b * LC + l * 32;
  float mx = -3.4e38f;
  for (int c = 0; c < 32; ++c) mx = fmaxf(mx, lg[c]);
  float se = 0.f;
  for (int c = 0; c < 32; ++c) se += __expf(lg[c] - mx);
  float lse = mx + __logf(se);
  float best = -3.4e38f; int bc = 0;
  for (int c = 0; c < 32; ++c) {
    float logp = lg[c] - lse;
    float mix  = __logf(0.99f * __expf(logp) + 0.01f / 32.f);
    float g    = -__logf(-__logf(u[c] + 1e-20f) + 1e-20f);
    float y    = mix + g;
    if (y > best) { best = y; bc = c; }
  }
  for (int c = 0; c < 32; ++c) {
    float oh = (c == bc) ? 1.f : 0.f;
    out_stoch[(size_t)b * FEAT + l * 32 + c] = oh;
    xbf[(size_t)b * IN_P + l * 32 + c] = (c == bc) ? (unsigned short)0x3F80 : (unsigned short)0;
  }
}

// ---------------------------------------------------------------------------
extern "C" void kernel_launch(void* const* d_in, const int* in_sizes, int n_in,
                              void* d_out, int out_size, void* d_ws, size_t ws_size,
                              hipStream_t stream) {
  const float* stoch0  = (const float*)d_in[0];
  const float* deter0  = (const float*)d_in[1];
  const float* W_ih    = (const float*)d_in[2];
  const float* W_hh    = (const float*)d_in[3];
  const float* b_ih    = (const float*)d_in[4];
  const float* b_hh    = (const float*)d_in[5];
  const float* w1      = (const float*)d_in[6];
  const float* b1      = (const float*)d_in[7];
  const float* rms_w   = (const float*)d_in[8];
  const float* w2      = (const float*)d_in[9];
  const float* b2      = (const float*)d_in[10];
  const float* unif    = (const float*)d_in[11];
  const int*   actions = (const int*)d_in[12];
  float* out = (float*)d_out;

  size_t off = 0;
  char* wsb = (char*)d_ws;
  auto take = [&](size_t bytes) { char* p = wsb + off; off += (bytes + 255) & ~(size_t)255; return p; };
  unsigned short* wihb  = (unsigned short*)take((size_t)NB * G3 * IN_P * 2);
  unsigned short* whhb  = (unsigned short*)take((size_t)NB * G3 * BS * 2);
  unsigned short* w1t   = (unsigned short*)take((size_t)HID * D_ * 2);
  unsigned short* w2t   = (unsigned short*)take((size_t)LC * HID * 2);
  unsigned short* xbf   = (unsigned short*)take((size_t)B_ * IN_P * 2);
  unsigned short* hbfA  = (unsigned short*)take((size_t)B_ * D_ * 2);
  unsigned short* hbfB  = (unsigned short*)take((size_t)B_ * D_ * 2);
  float*          deter = (float*)take((size_t)B_ * D_ * 4);
  unsigned short* hidbf = (unsigned short*)take((size_t)B_ * HID * 2);
  float*          logit = (float*)take((size_t)B_ * LC * 4);

  auto blocks = [](size_t n) { return (unsigned)((n + 255) / 256); };

  // one-time per launch: weight conversion (L2-resident afterwards) + state init
  k_conv_pad<<<blocks((size_t)NB*G3*IN_P), 256, 0, stream>>>(W_ih, wihb, NB*G3, 1030, IN_P);
  k_conv_pad<<<blocks((size_t)NB*G3*BS),   256, 0, stream>>>(W_hh, whhb, NB*G3, BS, BS);
  k_conv_t  <<<blocks((size_t)HID*D_),     256, 0, stream>>>(w1, w1t, D_, HID);
  k_conv_t  <<<blocks((size_t)LC*HID),     256, 0, stream>>>(w2, w2t, HID, LC);
  k_init_x  <<<blocks((size_t)B_*IN_P),    256, 0, stream>>>(stoch0, xbf);
  k_init_h  <<<blocks((size_t)B_*D_),      256, 0, stream>>>(deter0, deter, hbfA);

  unsigned short* hin = hbfA;
  unsigned short* hout = hbfB;
  for (int t = 0; t < T_; ++t) {
    float* feat_t = out + (size_t)t * B_ * FEAT;
    k_setact<<<4, 256, 0, stream>>>(actions + (size_t)t * B_, xbf);
    k_gates_gru<<<dim3(B_/32, NB, BS/128), 256, 0, stream>>>(
        xbf, hin, wihb, whhb, b_ih, b_hh, deter, hout, feat_t);
    k_mlp1<<<B_/16, 256, 0, stream>>>(hout, w1t, b1, rms_w, hidbf);
    k_logits<<<dim3(B_/32, LC/512), 256, 0, stream>>>(hidbf, w2t, b2, logit);
    k_sample<<<(B_*32)/256, 256, 0, stream>>>(logit, unif + (size_t)t * B_ * LC,
                                              feat_t + D_, xbf);
    unsigned short* tmp = hin; hin = hout; hout = tmp;
  }
}
